// HeadedToeplitzCausalLinear_15934328668570
// MI455X (gfx1250) — matrix operand order repaired and checked by reading, not compiled, not run on hardware
//
#include <hip/hip_runtime.h>

// HeadedToeplitzCausalLinear on gfx1250 (MI455X):
//   out[b,n,k] = sum_{m<=k} x[b,n,m] * weight[b%16, k-m] + bias[b%16, k]
//
// v_wmma_f32_16x16x32_bf16, split-bf16 (hi+lo) fp32 emulation, transposed
// tiles D' = Wt_tile (A op) x Xt_tile (B op); both operands are aligned
// ds_load_b128, zero unpack VALU. Each wave carries a PAIR of adjacent
// n-tiles (2p,2p+1): shared ktmax == p, shared x (B) fragments, and the
// second tile's Toeplitz run = same base shifted -16 elems. hi/lo planes are
// INTERLEAVED per row/replica so each fragment stream uses ONE bumped base
// register + small immediate DS offsets (Wall declared first so even folded
// constants fit the 16-bit DS offset; dynamic offsets laundered through
// inline asm to keep the base in a VGPR). Inner loop: ~10 unique
// ds_load_b128 + 6 WMMAs on 6 independent accumulator chains.

#define HEADS    16
#define NSEQ     64
#define MDIM     2048
#define ROW_TILE 16

#define XLOFF    2048   // lo plane offset within an x row (elems, 4096B)
#define XSTR     4120   // x row stride in elems (8240B == 48 mod 256 -> bank spread)
#define WCOPY    2120   // valid elems per weight replica plane
#define WLOFF    2120   // lo plane offset within a replica (elems, 4240B)
#define WSTR     4248   // replica stride in elems (8496B == 48 mod 256)
#define NREP     8
#define JSPLIT   2

typedef __attribute__((ext_vector_type(16))) __bf16 v16bf;
typedef __attribute__((ext_vector_type(8)))  float  v8f;

union VB { uint4 q[2]; v16bf v; };
union VF { float f[8]; v8f  v; };

__device__ __forceinline__ unsigned short bf16_rne(float x) {
    const unsigned int b = __float_as_uint(x);
    return (unsigned short)((b + 0x7fffu + ((b >> 16) & 1u)) >> 16);
}
__device__ __forceinline__ void split_bf16(float x, unsigned short& hi, unsigned short& lo) {
    hi = bf16_rne(x);
    const float hif = __uint_as_float((unsigned int)hi << 16);
    lo = bf16_rne(x - hif);
}

__device__ __forceinline__ v8f wmma_bf16(const VB& a, const VB& b, v8f c) {
    return __builtin_amdgcn_wmma_f32_16x16x32_bf16(
        false, a.v, false, b.v, (short)0, c, false, false);
}

__global__ __launch_bounds__(256)
void toeplitz_wmma_kernel(const float* __restrict__ x,
                          const float* __restrict__ weight,
                          const float* __restrict__ bias,
                          float* __restrict__ out)
{
    // Wall FIRST: its (possibly constant-folded) DS offsets stay < 64KB.
    __shared__ __align__(16) unsigned short Wall[NREP * WSTR];      //  67,968B
    __shared__ __align__(16) unsigned short Xall[ROW_TILE * XSTR];  // 131,840B

    const int h      = blockIdx.x;      // head
    const int rb     = blockIdx.y;      // 16-row block within head
    const int zsel   = blockIdx.z;      // pair-index interleave
    const int tid    = threadIdx.x;
    const int lane   = tid & 31;
    const int wave_u = __builtin_amdgcn_readfirstlane(tid >> 5);  // uniform SGPR
    const int nn     = lane & 15;
    const int half   = lane >> 4;
    const int m0     = rb * ROW_TILE;

    // ---- stage reversed weight, 8 shift-replicas, hi/lo planes ----
    for (int c = 0; c < NREP; ++c) {
        for (int i = tid; i < WCOPY; i += 256) {
            const int rr = i - c;                     // element index in wr[]
            float v = 0.0f;
            if (rr >= 0 && rr < MDIM) v = weight[h * MDIM + (MDIM - 1 - rr)];
            unsigned short vh, vl;
            split_bf16(v, vh, vl);
            Wall[c * WSTR + i]         = vh;
            Wall[c * WSTR + WLOFF + i] = vl;
        }
    }

    // ---- stage x panel: 16 rows x 2048, hi/lo planes per row ----
    for (int e = tid * 4; e < ROW_TILE * MDIM; e += 1024) {
        const int row  = e >> 11;
        const int col  = e & (MDIM - 1);
        const int grow = m0 + row;
        const int bidx = (grow >> 6) * HEADS + h;     // b = r*16 + h
        const int nidx = grow & (NSEQ - 1);
        const float4 v4 = *reinterpret_cast<const float4*>(
            x + (size_t)(bidx * NSEQ + nidx) * MDIM + col);
        unsigned short h0, l0, h1, l1, h2, l2, h3, l3;
        split_bf16(v4.x, h0, l0);
        split_bf16(v4.y, h1, l1);
        split_bf16(v4.z, h2, l2);
        split_bf16(v4.w, h3, l3);
        uint2 ph, pl;
        ph.x = (unsigned int)h0 | ((unsigned int)h1 << 16);
        ph.y = (unsigned int)h2 | ((unsigned int)h3 << 16);
        pl.x = (unsigned int)l0 | ((unsigned int)l1 << 16);
        pl.y = (unsigned int)l2 | ((unsigned int)l3 << 16);
        *reinterpret_cast<uint2*>(&Xall[row * XSTR + col])         = ph;
        *reinterpret_cast<uint2*>(&Xall[row * XSTR + XLOFF + col]) = pl;
    }
    __syncthreads();

    // ---- per-wave tile pairs (2 adjacent 16-col n-tiles, shared k-loop) ----
    #pragma unroll 1
    for (int jj = 0; jj < 4; ++jj) {
        const int p  = jj * 16 + wave_u * 2 + zsel;   // pair index 0..63 (uniform)
        const int j0 = p * 32;                        // first col of the pair
        const int ncol  = j0 + nn;                    // tile0 column for this lane
        const int ktmax = p;                          // shared, inclusive, SGPR

        // Toeplitz run for tile0 starts at wr index r1 (+k0); tile1 at r1-16.
        // r1 >= 16 always (ncol <= 2031 for tile0), so base at r1-16 is safe.
        const int r1 = (MDIM - 1) - ncol + 8 * half;
        const int c  = (-r1) & (NREP - 1);
        int aoff = c * WSTR + r1 + c - 16;
        int boff = nn * XSTR + 8 * half;
        // Launder the dynamic offsets: keeps the computed base in a VGPR so
        // in-loop accesses become base + small immediate (ds offset field),
        // without disturbing LDS addrspace inference.
        asm volatile("" : "+v"(aoff));
        asm volatile("" : "+v"(boff));
        const unsigned short* pA = &Wall[aoff];
        const unsigned short* pB = &Xall[boff];

        v8f a00 = {}, a01 = {}, a02 = {};             // tile0: hh, hl, lh
        v8f a10 = {}, a11 = {}, a12 = {};             // tile1: hh, hl, lh
        for (int kt = 0; kt <= ktmax; ++kt) {
            VB bh, bl, ah0, al0, ah1, al1;
            bh.q[0]  = *reinterpret_cast<const uint4*>(pB);
            bh.q[1]  = *reinterpret_cast<const uint4*>(pB + 16);
            bl.q[0]  = *reinterpret_cast<const uint4*>(pB + XLOFF);
            bl.q[1]  = *reinterpret_cast<const uint4*>(pB + XLOFF + 16);
            ah1.q[0] = *reinterpret_cast<const uint4*>(pA);
            ah1.q[1] = *reinterpret_cast<const uint4*>(pA + 16);
            ah0.q[0] = *reinterpret_cast<const uint4*>(pA + 16);   // CSE w/ ah1.q[1]
            ah0.q[1] = *reinterpret_cast<const uint4*>(pA + 32);
            al1.q[0] = *reinterpret_cast<const uint4*>(pA + WLOFF);
            al1.q[1] = *reinterpret_cast<const uint4*>(pA + WLOFF + 16);
            al0.q[0] = *reinterpret_cast<const uint4*>(pA + WLOFF + 16);
            al0.q[1] = *reinterpret_cast<const uint4*>(pA + WLOFF + 32);

            a00 = wmma_bf16(ah0, bh, a00);
            a01 = wmma_bf16(ah0, bl, a01);
            a02 = wmma_bf16(al0, bh, a02);
            a10 = wmma_bf16(ah1, bh, a10);
            a11 = wmma_bf16(ah1, bl, a11);
            a12 = wmma_bf16(al1, bh, a12);

            pA += 32;
            pB += 32;
        }

        // Epilogue: lane owns 8 contiguous output cols per tile of one row.
        const int grow = m0 + nn;
        const int bidx = (grow >> 6) * HEADS + h;
        const int nidx = grow & (NSEQ - 1);
        float* orow = out + (size_t)(bidx * NSEQ + nidx) * MDIM;

        #pragma unroll
        for (int t = 0; t < 2; ++t) {
            const int cb = j0 + 16 * t + 8 * half;    // first of 8 cols
            const float* bp = bias + h * MDIM + cb;
            const float4 ba = *reinterpret_cast<const float4*>(bp);
            const float4 bb = *reinterpret_cast<const float4*>(bp + 4);
            VF r0, r1v, r2;
            if (t == 0) { r0.v = a00; r1v.v = a01; r2.v = a02; }
            else        { r0.v = a10; r1v.v = a11; r2.v = a12; }
            float4 o0, o1;
            o0.x = r0.f[0] + r1v.f[0] + r2.f[0] + ba.x;
            o0.y = r0.f[1] + r1v.f[1] + r2.f[1] + ba.y;
            o0.z = r0.f[2] + r1v.f[2] + r2.f[2] + ba.z;
            o0.w = r0.f[3] + r1v.f[3] + r2.f[3] + ba.w;
            o1.x = r0.f[4] + r1v.f[4] + r2.f[4] + bb.x;
            o1.y = r0.f[5] + r1v.f[5] + r2.f[5] + bb.y;
            o1.z = r0.f[6] + r1v.f[6] + r2.f[6] + bb.z;
            o1.w = r0.f[7] + r1v.f[7] + r2.f[7] + bb.w;
            *reinterpret_cast<float4*>(orow + cb)     = o0;
            *reinterpret_cast<float4*>(orow + cb + 4) = o1;
        }
    }
}

extern "C" void kernel_launch(void* const* d_in, const int* in_sizes, int n_in,
                              void* d_out, int out_size, void* d_ws, size_t ws_size,
                              hipStream_t stream) {
    (void)in_sizes; (void)n_in; (void)out_size; (void)d_ws; (void)ws_size;
    const float* x = (const float*)d_in[0];
    const float* w = (const float*)d_in[1];
    const float* b = (const float*)d_in[2];
    float* out     = (float*)d_out;

    dim3 grid(HEADS, (4 * NSEQ) / ROW_TILE /* 16 row-blocks */, JSPLIT);
    toeplitz_wmma_kernel<<<grid, 256, 0, stream>>>(x, w, b, out);
}
